// Model_1211180778035
// MI455X (gfx1250) — matrix operand order repaired and checked by reading, not compiled, not run on hardware
//
#include <hip/hip_runtime.h>
#include <hip/hip_bf16.h>
#include <math.h>

// ---------------------------------------------------------------------------
// Problem constants (from reference)
// ---------------------------------------------------------------------------
#define BB     32
#define SS     512
#define EE     8
#define FF     16
#define KK     5
#define WW     492            // S - 20
#define MM     (BB * WW)      // 15744  (== 123 * 128)
#define DIN    1280
#define DPROJ  640
#define HH     128
#define G4     512            // 4*H
#define LL     4
#define LABEL  24

typedef __attribute__((ext_vector_type(16))) _Float16 v16h;
typedef __attribute__((ext_vector_type(8)))  _Float16 v8h;
typedef __attribute__((ext_vector_type(8)))  float    v8f;
typedef __attribute__((ext_vector_type(4)))  unsigned int u32x4;
typedef __attribute__((ext_vector_type(8)))  int      i32x8;
typedef __attribute__((ext_vector_type(4)))  int      i32x4;

#if defined(__gfx1250__) && __has_builtin(__builtin_amdgcn_tensor_load_to_lds)
#define USE_TDM 1
#else
#define USE_TDM 0
#endif

__device__ __forceinline__ float sigmoidf_(float x) {
    return 1.0f / (1.0f + __expf(-x));
}

// Build a 16-half WMMA operand register from two 16-byte chunks.
__device__ __forceinline__ v16h frag_from(const _Float16* p_lo, const _Float16* p_hi) {
    v16h r;
    ((v8h*)&r)[0] = *(const v8h*)p_lo;
    ((v8h*)&r)[1] = *(const v8h*)p_hi;
    return r;
}

#if USE_TDM
// ---------------------------------------------------------------------------
// TDM: async 2-D tile copy global -> LDS (f16 elements, data_size = 2B).
// D# layout per CDNA5 ISA 8.3/8.4: group0 = {flags, lds_addr, global_addr,
// type=2}; group1 = {mask/data_size, tensor dims, tile dims, strides}.
// Groups 2/3 unused for 2-D tiles (zeros).
// ---------------------------------------------------------------------------
__device__ __forceinline__ void tdm_load_tile_f16(
    unsigned int lds_off_bytes, const _Float16* gsrc,
    unsigned int tile_d0, unsigned int tile_d1,       // elements, rows
    unsigned long long tensor_d0, unsigned long long tensor_d1,
    unsigned long long row_stride_elems) {
    unsigned long long ga = (unsigned long long)(uintptr_t)gsrc;
    u32x4 g0;
    g0[0] = 1u;                                           // count=1, user mode
    g0[1] = lds_off_bytes;                                // lds_addr
    g0[2] = (unsigned int)(ga & 0xFFFFFFFFu);             // global_addr[31:0]
    g0[3] = (unsigned int)((ga >> 32) & 0x1FFFFFFu)       // global_addr[56:32]
          | (2u << 30);                                   // type = 2 (image)
    i32x8 g1;
    g1[0] = (int)(1u << 16);                              // data_size=1 (2B), wg_mask=0
    g1[1] = (int)((tensor_d0 & 0xFFFFu) << 16);           // tensor_dim0[15:0]
    g1[2] = (int)(((tensor_d0 >> 16) & 0xFFFFu)
          | ((tensor_d1 & 0xFFFFu) << 16));               // td0[31:16] | td1[15:0]
    g1[3] = (int)(((tensor_d1 >> 16) & 0xFFFFu)
          | ((tile_d0 & 0xFFFFu) << 16));                 // td1[31:16] | tile_dim0
    g1[4] = (int)(tile_d1 & 0xFFFFu);                     // tile_dim1, tile_dim2=0
    g1[5] = (int)(unsigned int)(row_stride_elems & 0xFFFFFFFFu);  // stride0[31:0]
    g1[6] = (int)(unsigned int)((row_stride_elems >> 32) & 0xFFFFu); // stride0[47:32]
    g1[7] = 0;
    i32x4 z4 = {0, 0, 0, 0};
#if defined(__clang_major__) && (__clang_major__ >= 23)
    i32x8 z8 = {0, 0, 0, 0, 0, 0, 0, 0};
    __builtin_amdgcn_tensor_load_to_lds(g0, g1, z4, z4, z8, 0);
#else
    __builtin_amdgcn_tensor_load_to_lds(g0, g1, z4, z4, 0);
#endif
}
#endif // USE_TDM

// ---------------------------------------------------------------------------
// 1) f32 -> f16 conversion
// ---------------------------------------------------------------------------
__global__ void cvt_f32_to_f16(const float* __restrict__ in,
                               _Float16* __restrict__ out, int n) {
    int i = blockIdx.x * 256 + threadIdx.x;
    if (i < n) out[i] = (_Float16)in[i];
}

// ---------------------------------------------------------------------------
// 2) Fused sliding-window conv1d(SAME,K=5) + bias + MaxPool1d(2) -> f16
// ---------------------------------------------------------------------------
__global__ __launch_bounds__(256)
void conv_pool_kernel(const float* __restrict__ in,     // (32,512,8)
                      const float* __restrict__ cw,     // (16,1,5)
                      const float* __restrict__ cb,     // (16,)
                      _Float16* __restrict__ pooled) {  // (MM, DIN) f16
    __shared__ float row[SS * EE];       // 16 KB
    __shared__ float wsh[FF][KK];
    __shared__ float bsh[FF];
    const int b = blockIdx.x;
    for (int i = threadIdx.x; i < SS * EE; i += 256) row[i] = in[b * SS * EE + i];
    if (threadIdx.x < FF * KK) wsh[threadIdx.x / KK][threadIdx.x % KK] = cw[threadIdx.x];
    if (threadIdx.x < FF) bsh[threadIdx.x] = cb[threadIdx.x];
    __syncthreads();

    const int TOT = WW * DIN;
    for (int idx = threadIdx.x; idx < TOT; idx += 256) {
        const int w  = idx / DIN;
        const int kk = idx - w * DIN;
        const int f  = kk / 80;
        const int p  = kk - f * 80;
        const int base = w * EE;         // window = row[base .. base+160)
        float m = -3.402823466e38f;
        #pragma unroll
        for (int q = 0; q < 2; ++q) {
            const int pos = 2 * p + q;
            float s = bsh[f];
            #pragma unroll
            for (int t = 0; t < KK; ++t) {
                const int x = pos + t - 2;
                const float v = (x >= 0 && x < 160) ? row[base + x] : 0.0f;
                s = fmaf(v, wsh[f][t], s);
            }
            m = fmaxf(m, s);
        }
        __builtin_nontemporal_store((_Float16)m,
            pooled + (size_t)(b * WW + w) * DIN + kk);
    }
}

// ---------------------------------------------------------------------------
// 3) WMMA GEMM with TDM double-buffered LDS staging.
//    out[M,N] = A[M,K](f16) @ W[N,K]^T(f16) + bias[N]
//    Block = 256 threads = 8 waves (4 along M x 2 along N); tile 128x64.
//    Per K-step (32): A tile 128x32 (8KB) + B tile 64x32 (4KB) in LDS,
//    double buffered; wave 0 drives the TDM, barriers publish completion.
//    Requires M%128==0, N%64==0, K%32==0 (true for all calls here).
// ---------------------------------------------------------------------------
template<bool OUT_F16>
__global__ __launch_bounds__(256)
void wmma_gemm_kernel(const _Float16* __restrict__ A,
                      const _Float16* __restrict__ Wt,   // (N,K) row-major
                      const float*    __restrict__ bias, // (N,)
                      void*           __restrict__ outp,
                      int M, int N, int K) {
    __shared__ _Float16 Abuf[2][128 * 32];   // 2 x 8KB
    __shared__ _Float16 Bbuf[2][64 * 32];    // 2 x 4KB

    const int lane  = threadIdx.x & 31;
    const int wave  = threadIdx.x >> 5;    // 0..7
    const int waveM = wave >> 1;           // 0..3
    const int waveN = wave & 1;            // 0..1
    const int lr    = lane & 15;
    const int lh    = lane >> 4;           // 0/1

    const int mBase = blockIdx.x * 128;
    const int nBase = blockIdx.y * 64;
    const int mBlk  = mBase + waveM * 32;
    const int nBlk  = nBase + waveN * 32;
    const int nk    = K >> 5;              // K/32 steps

#if USE_TDM
    const bool issuer = (threadIdx.x < 32);   // wave 0 drives the TDM
    if (issuer) {
        tdm_load_tile_f16((unsigned int)(uintptr_t)&Abuf[0][0],
                          A + (size_t)mBase * K, 32, 128,
                          (unsigned long long)K, (unsigned long long)M,
                          (unsigned long long)K);
        tdm_load_tile_f16((unsigned int)(uintptr_t)&Bbuf[0][0],
                          Wt + (size_t)nBase * K, 32, 64,
                          (unsigned long long)K, (unsigned long long)N,
                          (unsigned long long)K);
    }
#endif

    v8f acc[2][2] = {};

    for (int ks = 0; ks < nk; ++ks) {
        const int cur = ks & 1;
#if USE_TDM
        const bool issuer2 = (threadIdx.x < 32);
        if (issuer2) __builtin_amdgcn_s_wait_tensorcnt((short)0);
        __syncthreads();                         // buf[cur] ready for everyone
        if (issuer2 && (ks + 1 < nk)) {
            const int kk = (ks + 1) << 5;
            tdm_load_tile_f16((unsigned int)(uintptr_t)&Abuf[1 - cur][0],
                              A + (size_t)mBase * K + kk, 32, 128,
                              (unsigned long long)K, (unsigned long long)M,
                              (unsigned long long)K);
            tdm_load_tile_f16((unsigned int)(uintptr_t)&Bbuf[1 - cur][0],
                              Wt + (size_t)nBase * K + kk, 32, 64,
                              (unsigned long long)K, (unsigned long long)N,
                              (unsigned long long)K);
        }
#else
        // Fallback staging: cooperative vectorized copy (v8h chunks).
        {
            const int kk = ks << 5;
            // A tile: 128 rows x 32 halves = 512 v8h chunks
            for (int c = threadIdx.x; c < 512; c += 256) {
                const int r = c >> 2, seg = c & 3;
                *(v8h*)&Abuf[cur][r * 32 + seg * 8] =
                    *(const v8h*)(A + (size_t)(mBase + r) * K + kk + seg * 8);
            }
            // B tile: 64 rows x 32 halves = 256 v8h chunks
            {
                const int c = threadIdx.x;
                const int r = c >> 2, seg = c & 3;
                *(v8h*)&Bbuf[cur][r * 32 + seg * 8] =
                    *(const v8h*)(Wt + (size_t)(nBase + r) * K + kk + seg * 8);
            }
            __syncthreads();
        }
#endif
        // --- compute from LDS buf[cur]
        v16h a[2], b[2];
        #pragma unroll
        for (int i = 0; i < 2; ++i) {
            const _Float16* pa = &Abuf[cur][(waveM * 32 + i * 16 + lr) * 32 + lh * 8];
            a[i] = frag_from(pa, pa + 16);
            const _Float16* pb = &Bbuf[cur][(waveN * 32 + i * 16 + lr) * 32 + lh * 8];
            b[i] = frag_from(pb, pb + 16);
        }
        #pragma unroll
        for (int i = 0; i < 2; ++i)
            #pragma unroll
            for (int j = 0; j < 2; ++j)
                acc[i][j] = __builtin_amdgcn_wmma_f32_16x16x32_f16(
                    false, a[i], false, b[j], (short)0, acc[i][j], false, false);
        __syncthreads();                         // reads of buf[cur] complete
    }

    #pragma unroll
    for (int j = 0; j < 2; ++j) {
        const float bv = bias ? bias[nBlk + j * 16 + lr] : 0.0f;
        #pragma unroll
        for (int i = 0; i < 2; ++i) {
            #pragma unroll
            for (int r = 0; r < 8; ++r) {
                const int m = mBlk + i * 16 + r + lh * 8;
                const int n = nBlk + j * 16 + lr;
                const float v = acc[i][j][r] + bv;
                if (OUT_F16)
                    __builtin_nontemporal_store((_Float16)v,
                        (_Float16*)outp + (size_t)m * N + n);
                else
                    __builtin_nontemporal_store(v,
                        (float*)outp + (size_t)m * N + n);
            }
        }
    }
}

// ---------------------------------------------------------------------------
// 4) LSTM recurrence for one layer (persistent single block, 8 waves).
//    Per step: G(32x512) = h_f16(32x128) @ w_hh^T via WMMA (A from LDS),
//    then gates = G + xg[b,t,:] + b_hh ; elementwise LSTM update.
//    Dynamic LDS: G f16 (32KB) + h f16 (8KB) + c f32 (16KB) = 56KB.
// ---------------------------------------------------------------------------
__global__ __launch_bounds__(256)
void lstm_scan_kernel(const float*    __restrict__ xg,    // (MM, 512), row b*WW+t
                      const _Float16* __restrict__ whh,   // (512,128) f16
                      const float*    __restrict__ bhh,   // (512,)
                      _Float16*       __restrict__ hs,    // (MM, 128) f16 out
                      float*          __restrict__ hfin)  // (32,128) f32 out
{
    extern __shared__ char smem[];
    _Float16* Gs = (_Float16*)smem;                            // 32*512 f16
    _Float16* hF = (_Float16*)(smem + BB * G4 * 2);            // 32*128 f16
    float*    cS = (float*)(smem + BB * G4 * 2 + BB * HH * 2); // 32*128 f32

    for (int i = threadIdx.x; i < BB * HH; i += 256) {
        hF[i] = (_Float16)0.0f;
        cS[i] = 0.0f;
    }
    __syncthreads();

    const int lane = threadIdx.x & 31;
    const int wave = threadIdx.x >> 5;
    const int lr   = lane & 15;
    const int lh   = lane >> 4;

    for (int t = 0; t < WW; ++t) {
        // --- gate GEMM: each wave covers N in [wave*64, wave*64+64), both M tiles
        #pragma unroll
        for (int jn = 0; jn < 4; ++jn) {
            const int n0 = wave * 64 + jn * 16;
            v8f acc0 = {}, acc1 = {};
            #pragma unroll
            for (int k = 0; k < HH; k += 32) {
                const _Float16* pb = whh + (size_t)(n0 + lr) * HH + k + lh * 8;
                const v16h bfrag = frag_from(pb, pb + 16);
                const _Float16* pa0 = hF + (0  + lr) * HH + k + lh * 8;  // LDS
                const _Float16* pa1 = hF + (16 + lr) * HH + k + lh * 8;  // LDS
                const v16h a0 = frag_from(pa0, pa0 + 16);
                const v16h a1 = frag_from(pa1, pa1 + 16);
                acc0 = __builtin_amdgcn_wmma_f32_16x16x32_f16(
                    false, a0, false, bfrag, (short)0, acc0, false, false);
                acc1 = __builtin_amdgcn_wmma_f32_16x16x32_f16(
                    false, a1, false, bfrag, (short)0, acc1, false, false);
            }
            #pragma unroll
            for (int r = 0; r < 8; ++r) {
                Gs[(size_t)(0  + r + lh * 8) * G4 + n0 + lr] = (_Float16)acc0[r];
                Gs[(size_t)(16 + r + lh * 8) * G4 + n0 + lr] = (_Float16)acc1[r];
            }
        }
        __syncthreads();

        // --- elementwise LSTM cell update (gate order: i, f, g, o)
        for (int u = threadIdx.x; u < BB * HH; u += 256) {
            const int b = u >> 7;        // u / 128
            const int j = u & (HH - 1);  // u % 128
            const float* xr = xg + ((size_t)b * WW + t) * G4;
            const float gi = (float)Gs[(size_t)b * G4 + j]          + xr[j]          + bhh[j];
            const float gf = (float)Gs[(size_t)b * G4 + HH + j]     + xr[HH + j]     + bhh[HH + j];
            const float gg = (float)Gs[(size_t)b * G4 + 2 * HH + j] + xr[2 * HH + j] + bhh[2 * HH + j];
            const float go = (float)Gs[(size_t)b * G4 + 3 * HH + j] + xr[3 * HH + j] + bhh[3 * HH + j];
            const float c  = sigmoidf_(gf) * cS[u] + sigmoidf_(gi) * tanhf(gg);
            const float h  = sigmoidf_(go) * tanhf(c);
            cS[u] = c;
            hF[u] = (_Float16)h;
            __builtin_nontemporal_store((_Float16)h,
                hs + ((size_t)b * WW + t) * HH + j);
            if (t == WW - 1) hfin[u] = h;
        }
        __syncthreads();
    }
}

// ---------------------------------------------------------------------------
// 5) Final Linear: out[b,o] = h[b,:] . out_w[o,:] + out_b[o]   (32 x 24)
// ---------------------------------------------------------------------------
__global__ void final_out_kernel(const float* __restrict__ h,
                                 const float* __restrict__ ow,
                                 const float* __restrict__ ob,
                                 float* __restrict__ out) {
    const int idx = blockIdx.x * 256 + threadIdx.x;
    if (idx >= BB * LABEL) return;
    const int b = idx / LABEL;
    const int o = idx - b * LABEL;
    float s = ob[o];
    #pragma unroll 4
    for (int j = 0; j < HH; ++j) s = fmaf(h[b * HH + j], ow[o * HH + j], s);
    out[idx] = s;
}

// ---------------------------------------------------------------------------
// Host-side orchestration
// ---------------------------------------------------------------------------
extern "C" void kernel_launch(void* const* d_in, const int* in_sizes, int n_in,
                              void* d_out, int out_size, void* d_ws, size_t ws_size,
                              hipStream_t stream) {
    // 0 inputs, 1 _x, 2 y, 3 _y, 4 conv_w, 5 conv_b, 6 proj_w, 7 proj_b,
    // 8 w_ih0, 9 w_ih_rest, 10 w_hh, 11 b_ih, 12 b_hh, 13 out_w, 14 out_b
    const float* inputs = (const float*)d_in[0];
    const float* conv_w = (const float*)d_in[4];
    const float* conv_b = (const float*)d_in[5];
    const float* proj_w = (const float*)d_in[6];
    const float* proj_b = (const float*)d_in[7];
    const float* w_ih0  = (const float*)d_in[8];
    const float* w_ihR  = (const float*)d_in[9];
    const float* w_hh   = (const float*)d_in[10];
    const float* b_ih   = (const float*)d_in[11];
    const float* b_hh   = (const float*)d_in[12];
    const float* out_w  = (const float*)d_in[13];
    const float* out_b  = (const float*)d_in[14];
    float* out = (float*)d_out;

    // Workspace carve-up (256B aligned)
    char* ws = (char*)d_ws;
    size_t off = 0;
    auto carve = [&](size_t bytes) -> char* {
        char* p = ws + off;
        off = (off + bytes + 255) & ~(size_t)255;
        return p;
    };
    _Float16* pooled16 = (_Float16*)carve((size_t)MM * DIN * 2);
    _Float16* seq16    = (_Float16*)carve((size_t)MM * DPROJ * 2);
    float*    xg       = (float*)   carve((size_t)MM * G4 * 4);
    _Float16* hsA      = (_Float16*)carve((size_t)MM * HH * 2);
    _Float16* hsB      = (_Float16*)carve((size_t)MM * HH * 2);
    float*    hfin     = (float*)   carve((size_t)BB * HH * 4);
    _Float16* projw16  = (_Float16*)carve((size_t)DPROJ * DIN * 2);
    _Float16* wih016   = (_Float16*)carve((size_t)G4 * DPROJ * 2);
    _Float16* wihR16   = (_Float16*)carve((size_t)(LL - 1) * G4 * HH * 2);
    _Float16* whh16    = (_Float16*)carve((size_t)LL * G4 * HH * 2);
    (void)ws_size; (void)n_in; (void)in_sizes; (void)out_size;

    // 1) weight conversion
    {
        const int n1 = DPROJ * DIN;
        cvt_f32_to_f16<<<(n1 + 255) / 256, 256, 0, stream>>>(proj_w, projw16, n1);
        const int n2 = G4 * DPROJ;
        cvt_f32_to_f16<<<(n2 + 255) / 256, 256, 0, stream>>>(w_ih0, wih016, n2);
        const int n3 = (LL - 1) * G4 * HH;
        cvt_f32_to_f16<<<(n3 + 255) / 256, 256, 0, stream>>>(w_ihR, wihR16, n3);
        const int n4 = LL * G4 * HH;
        cvt_f32_to_f16<<<(n4 + 255) / 256, 256, 0, stream>>>(w_hh, whh16, n4);
    }

    // 2) fused window + conv + maxpool
    conv_pool_kernel<<<BB, 256, 0, stream>>>(inputs, conv_w, conv_b, pooled16);

    // 3) projection GEMM: (MM x DIN) x (DPROJ x DIN)^T -> seq16 (f16)
    {
        dim3 grid(MM / 128, DPROJ / 64);
        wmma_gemm_kernel<true><<<grid, 256, 0, stream>>>(
            pooled16, projw16, proj_b, seq16, MM, DPROJ, DIN);
    }

    // 4) four LSTM layers: input-gate GEMM + recurrence scan
    const size_t scan_lds = (size_t)BB * G4 * 2 + (size_t)BB * HH * 2 + (size_t)BB * HH * 4;
    _Float16* hsPrev = hsA;
    _Float16* hsCur  = hsB;
    for (int l = 0; l < LL; ++l) {
        const _Float16* Ain = (l == 0) ? seq16 : hsPrev;
        const _Float16* Wl  = (l == 0) ? wih016 : (wihR16 + (size_t)(l - 1) * G4 * HH);
        const int Kl        = (l == 0) ? DPROJ : HH;
        dim3 grid(MM / 128, G4 / 64);
        wmma_gemm_kernel<false><<<grid, 256, 0, stream>>>(
            Ain, Wl, b_ih + (size_t)l * G4, xg, MM, G4, Kl);

        lstm_scan_kernel<<<1, 256, scan_lds, stream>>>(
            xg, whh16 + (size_t)l * G4 * HH, b_hh + (size_t)l * G4, hsCur, hfin);

        _Float16* tmp = hsPrev; hsPrev = hsCur; hsCur = tmp;
    }

    // 5) final linear
    final_out_kernel<<<(BB * LABEL + 255) / 256, 256, 0, stream>>>(
        hfin, out_w, out_b, out);
}
